// WindowAttention_52329881535100
// MI455X (gfx1250) — matrix operand order repaired and checked by reading, not compiled, not run on hardware
//
#include <hip/hip_runtime.h>

// -----------------------------------------------------------------------------
// Window attention, fused per batch element (b = blockIdx.x, 1024 blocks).
// All 4 sequential iterations run inside one block; feat/Q/K/V/P live in LDS
// (~318 KB of the WGP's 320 KB). GEMMs use v_wmma_f32_16x16x32_bf16.
// Shapes: B_=1024, N=256, C=512, nh=4, hd=128, n=64, nW=64.
//
// Roofline: ~585 GFLOP total vs ~1.1 GB HBM traffic -> compute bound at bf16
// WMMA rates; weights (4 MB f32) stay resident in the 192 MB L2.
// -----------------------------------------------------------------------------

#define C_DIM 512
#define NH    4
#define HD    128
#define NSUB  64
#define QK_SCALE 0.08838834764831845f  // 128^-0.5

typedef __attribute__((ext_vector_type(16))) __bf16          v16bf;
typedef __attribute__((ext_vector_type(8)))  float           v8f;
typedef __attribute__((ext_vector_type(8)))  unsigned short  us8;

// LDS layout (ushort elements = raw bf16 bits). Row pads (+8) avoid bank
// conflicts on the 16-lane strided A-fragment loads.
#define SA_LD 520   // feat / attn-out : 64 x 512
#define SQ_LD 136   // Q : [h][t][d]    4 x 64 x 128
#define SK_LD 72    // K : [h][d][t]    4 x 128 x 64  (pre-transposed -> B operand)
#define SV_LD 136   // V : [h][t][d]
#define SP_LD 72    // P : [h][t][t]
#define SW_LD 136   // weight staging [k][n] : 32 x 128

#define SA_OFF 0
#define SQ_OFF (SA_OFF + 64 * SA_LD)
#define SK_OFF (SQ_OFF + NH * 64 * SQ_LD)
#define SV_OFF (SK_OFF + NH * 128 * SK_LD)
#define SP_OFF (SV_OFF + NH * 64 * SV_LD)
#define SW_OFF (SP_OFF + NH * 64 * SP_LD)
#define SMEM_ELEMS (SW_OFF + 32 * SW_LD)   // 162560 elems
#define SMEM_BYTES (SMEM_ELEMS * 2)        // 325120 B  (<= 320 KB)

// combined bias+mask table in workspace: [w][h][q][k] = 64*4*64*64 f32 = 4 MB
#define BM_PER_W (NH * NSUB * NSUB)
#define BM_ELEMS (64 * BM_PER_W)

__device__ __forceinline__ unsigned short f2bf(float f) {
  unsigned u = __float_as_uint(f);
  u += 0x7fffu + ((u >> 16) & 1u);     // round-to-nearest-even
  return (unsigned short)(u >> 16);
}
__device__ __forceinline__ float bf2f(unsigned short h) {
  return __uint_as_float(((unsigned)h) << 16);
}

union FragU { v16bf v; us8 h[2]; };

__device__ __forceinline__ v8f zero8() {
  v8f z;
#pragma unroll
  for (int i = 0; i < 8; ++i) z[i] = 0.0f;
  return z;
}

// A-matrix 16x32 bf16 fragment from row-major LDS tile (ISA 7.12.2):
// lanes 0-15: row m=lane, K 0-7 (VGPR0-3) and K 16-23 (VGPR4-7)
// lanes 16-31: row m=lane-16, K 8-15 and K 24-31
__device__ __forceinline__ v16bf load_a_frag(const unsigned short* base, int ld, int lane) {
  int m  = lane & 15;
  int kh = (lane >> 4) * 8;
  const unsigned short* r = base + m * ld;
  FragU u;
  u.h[0] = *(const us8*)(r + kh);
  u.h[1] = *(const us8*)(r + 16 + kh);
  return u.v;
}

// B-matrix 32x16 bf16 fragment from [k][n] LDS tile: lane L holds column n=L%16,
// K 8h..8h+7 then K 16+8h..16+8h+7 (h = L/16), matching the A fragment.
__device__ __forceinline__ v16bf load_b_frag(const unsigned short* base, int ld, int lane) {
  const unsigned short* r = base + (lane & 15);
  int kh = (lane >> 4) * 8;
  FragU u;
#pragma unroll
  for (int j = 0; j < 8; ++j) {
    u.h[0][j] = r[(kh + j) * ld];
    u.h[1][j] = r[(16 + kh + j) * ld];
  }
  return u.v;
}

__device__ __forceinline__ v8f wmma_bf16(v16bf a, v16bf b, v8f c) {
  return __builtin_amdgcn_wmma_f32_16x16x32_bf16(false, a, false, b, (short)0, c,
                                                 false, false);
}

// Stage a 128(row) x 32(k) block of an f32 weight matrix (row-major, ld=512)
// into LDS transposed+bf16 as [k][n] for B-operand consumption. 256 threads.
// Prefetches the next K slice (global_prefetch_b8) to overlap with WMMA.
__device__ __forceinline__ void stage_w_tile(const float* __restrict__ W,
                                             int row_base, int k_base,
                                             unsigned short* SW, int tid) {
  int r  = tid >> 1;          // 0..127 : weight row (output column n)
  int kh = (tid & 1) * 16;    // 0 or 16 : k half
  const float* src = W + (size_t)(row_base + r) * C_DIM + k_base + kh;
  __builtin_prefetch(src + 32, 0, 0);   // speculative: next k slice
  float4 f0 = ((const float4*)src)[0];
  float4 f1 = ((const float4*)src)[1];
  float4 f2 = ((const float4*)src)[2];
  float4 f3 = ((const float4*)src)[3];
  float v[16] = {f0.x, f0.y, f0.z, f0.w, f1.x, f1.y, f1.z, f1.w,
                 f2.x, f2.y, f2.z, f2.w, f3.x, f3.y, f3.z, f3.w};
#pragma unroll
  for (int j = 0; j < 16; ++j)
    SW[(kh + j) * SW_LD + r] = f2bf(v[j]);
}

// bm[w][h][q][k] = lanm[lanm_index[q*256 + 4k]*4 + h] + mask[w][q][k]
__global__ void bias_mask_kernel(const float* __restrict__ lanm,
                                 const int* __restrict__ lanm_index,
                                 const float* __restrict__ mask,
                                 float* __restrict__ bm) {
  int idx = blockIdx.x * 256 + threadIdx.x;    // < 64*4*64*64
  int w = idx >> 14;
  int h = (idx >> 12) & 3;
  int q = (idx >> 6) & 63;
  int k = idx & 63;
  float bias = lanm[lanm_index[q * 256 + 4 * k] * NH + h];
  bm[idx] = bias + mask[(w * NSUB + q) * NSUB + k];
}

__global__ __launch_bounds__(256) void window_attn_kernel(
    const float* __restrict__ x,
    const float* __restrict__ qkv_w, const float* __restrict__ qkv_b,
    const float* __restrict__ proj_w, const float* __restrict__ proj_b,
    const float* __restrict__ bm, float* __restrict__ out) {
  extern __shared__ unsigned short smem[];
  unsigned short* SA = smem + SA_OFF;
  unsigned short* SQ = smem + SQ_OFF;
  unsigned short* SK = smem + SK_OFF;
  unsigned short* SV = smem + SV_OFF;
  unsigned short* SP = smem + SP_OFF;
  unsigned short* SW = smem + SW_OFF;

  const int tid  = threadIdx.x;
  const int wave = tid >> 5;
  const int lane = tid & 31;
  const int lh   = lane >> 4;   // half-wave (row +8 in C layout)
  const int ln   = lane & 15;   // column within 16-wide tile
  const int b    = blockIdx.x;

  const float* xb  = x + (size_t)b * 256 * C_DIM;
  const float* bmb = bm + (size_t)(b & 63) * BM_PER_W;

  // ---- feat = bf16(x[b, 0:64, :]) ----
  for (int e = tid; e < 64 * C_DIM / 4; e += 256) {
    int r = (e * 4) >> 9, c = (e * 4) & 511;
    float4 f = *(const float4*)(xb + r * C_DIM + c);
    unsigned short* p = SA + r * SA_LD + c;
    p[0] = f2bf(f.x); p[1] = f2bf(f.y); p[2] = f2bf(f.z); p[3] = f2bf(f.w);
  }
  __syncthreads();

  const int mt  = wave & 3;          // row tile  (M = 64 -> 4 tiles)
  const int nt0 = (wave >> 2) * 4;   // 4 n-tiles within a 128-col chunk

  for (int it = 0; it < 4; ++it) {
    // ============ Phase 1: QKV = feat @ qkv_w^T + qkv_b ============
    for (int nc = 0; nc < 12; ++nc) {           // 1536 cols / 128
      v8f acc[4] = {zero8(), zero8(), zero8(), zero8()};
      for (int kk = 0; kk < 16; ++kk) {         // K = 512 / 32
        __syncthreads();
        stage_w_tile(qkv_w, nc * 128, kk * 32, SW, tid);
        __syncthreads();
        v16bf a = load_a_frag(SA + mt * 16 * SA_LD + kk * 32, SA_LD, lane);
#pragma unroll
        for (int t = 0; t < 4; ++t) {
          v16bf bfr = load_b_frag(SW + (nt0 + t) * 16, SW_LD, lane);
          acc[t] = wmma_bf16(a, bfr, acc[t]);
        }
      }
#pragma unroll
      for (int t = 0; t < 4; ++t) {
        // Tile-base column is wave-uniform: force into an SGPR so the Q/K/V
        // section branch is a scalar branch, not an exec-mask dance.
        int colbase = __builtin_amdgcn_readfirstlane(nc * 128 + (nt0 + t) * 16);
        int s = colbase >> 9;            // 0:Q 1:K 2:V (uniform)
        int h = (colbase >> 7) & 3;      // head (uniform)
        int d = (colbase & 127) + ln;    // dim within head
        float bc = qkv_b[colbase + ln];
        if (s == 0) {
#pragma unroll
          for (int v = 0; v < 8; ++v) {
            int m = mt * 16 + v + 8 * lh;
            SQ[(h * 64 + m) * SQ_LD + d] = f2bf((acc[t][v] + bc) * QK_SCALE);
          }
        } else if (s == 1) {
#pragma unroll
          for (int v = 0; v < 8; ++v) {
            int m = mt * 16 + v + 8 * lh;
            SK[(h * 128 + d) * SK_LD + m] = f2bf(acc[t][v] + bc);   // K^T
          }
        } else {
#pragma unroll
          for (int v = 0; v < 8; ++v) {
            int m = mt * 16 + v + 8 * lh;
            SV[(h * 64 + m) * SV_LD + d] = f2bf(acc[t][v] + bc);
          }
        }
      }
    }
    __syncthreads();

    // ============ Phase 2: logits = Q@K^T + (bias+mask), softmax ============
    for (int rg = wave; rg < 16; rg += 8) {     // 4 heads x 4 row tiles
      int h = rg >> 2, qt = rg & 3;
      v8f acc[4] = {zero8(), zero8(), zero8(), zero8()};
      for (int dd = 0; dd < 4; ++dd) {          // K = hd 128 / 32
        v16bf a = load_a_frag(SQ + (h * 64 + qt * 16) * SQ_LD + dd * 32, SQ_LD, lane);
#pragma unroll
        for (int nt = 0; nt < 4; ++nt) {
          v16bf bfr = load_b_frag(SK + (h * 128 + dd * 32) * SK_LD + nt * 16, SK_LD, lane);
          acc[nt] = wmma_bf16(a, bfr, acc[nt]);
        }
      }
#pragma unroll
      for (int v = 0; v < 8; ++v) {
        int m = qt * 16 + v + 8 * lh;
        const float* bmrow = bmb + (h * NSUB + m) * NSUB + ln;
        float p[4];
        float rmax = -3.0e38f;
#pragma unroll
        for (int nt = 0; nt < 4; ++nt) {
          p[nt] = acc[nt][v] + bmrow[nt * 16];
          rmax = fmaxf(rmax, p[nt]);
        }
#pragma unroll
        for (int o = 1; o < 16; o <<= 1) rmax = fmaxf(rmax, __shfl_xor(rmax, o, 16));
        float ssum = 0.0f;
#pragma unroll
        for (int nt = 0; nt < 4; ++nt) { p[nt] = __expf(p[nt] - rmax); ssum += p[nt]; }
#pragma unroll
        for (int o = 1; o < 16; o <<= 1) ssum += __shfl_xor(ssum, o, 16);
        float rs = 1.0f / ssum;
#pragma unroll
        for (int nt = 0; nt < 4; ++nt)
          SP[(h * 64 + m) * SP_LD + nt * 16 + ln] = f2bf(p[nt] * rs);
      }
    }
    __syncthreads();

    // ============ Phase 3: out = P @ V  (into SA, per-head col blocks) ======
    for (int idx = wave; idx < 128; idx += 8) {   // 4h x 4mt x 8nt tiles
      int h = idx >> 5, qt = (idx >> 3) & 3, nt = idx & 7;
      v8f acc = zero8();
#pragma unroll
      for (int kk = 0; kk < 2; ++kk) {            // K = 64 / 32
        v16bf a   = load_a_frag(SP + (h * 64 + qt * 16) * SP_LD + kk * 32, SP_LD, lane);
        v16bf bfr = load_b_frag(SV + (h * 64 + kk * 32) * SV_LD + nt * 16, SV_LD, lane);
        acc = wmma_bf16(a, bfr, acc);
      }
#pragma unroll
      for (int v = 0; v < 8; ++v)
        SA[(qt * 16 + v + 8 * lh) * SA_LD + h * 128 + nt * 16 + ln] = f2bf(acc[v]);
    }
    __syncthreads();

    // ============ Phase 4: d_out chunk = out @ proj_w^T + proj_b ============
    float* outb = out + ((size_t)b * 256 + it * 64) * C_DIM;
    for (int nc = 0; nc < 4; ++nc) {             // 512 cols / 128
      v8f acc[4] = {zero8(), zero8(), zero8(), zero8()};
      for (int kk = 0; kk < 16; ++kk) {
        __syncthreads();
        stage_w_tile(proj_w, nc * 128, kk * 32, SW, tid);
        __syncthreads();
        v16bf a = load_a_frag(SA + mt * 16 * SA_LD + kk * 32, SA_LD, lane);
#pragma unroll
        for (int t = 0; t < 4; ++t) {
          v16bf bfr = load_b_frag(SW + (nt0 + t) * 16, SW_LD, lane);
          acc[t] = wmma_bf16(a, bfr, acc[t]);
        }
      }
#pragma unroll
      for (int t = 0; t < 4; ++t) {
        int col = nc * 128 + (nt0 + t) * 16 + ln;
        float pb = proj_b[col];
#pragma unroll
        for (int v = 0; v < 8; ++v) {
          int m = mt * 16 + v + 8 * lh;
          outb[(size_t)m * C_DIM + col] = acc[t][v] + pb;
        }
      }
    }

    // ============ Phase 5: feat = bf16(out + x[b, (it+1)*64 : , :]) =========
    if (it < 3) {
      __syncthreads();
      const float* xn = xb + (size_t)(it + 1) * 64 * C_DIM;
      for (int e = tid; e < 64 * C_DIM / 4; e += 256) {
        int r = (e * 4) >> 9, c = (e * 4) & 511;
        float4 f = *(const float4*)(xn + r * C_DIM + c);
        unsigned short* p = SA + r * SA_LD + c;
        p[0] = f2bf(bf2f(p[0]) + f.x);
        p[1] = f2bf(bf2f(p[1]) + f.y);
        p[2] = f2bf(bf2f(p[2]) + f.z);
        p[3] = f2bf(bf2f(p[3]) + f.w);
      }
      __syncthreads();
    }
  }
}

extern "C" void kernel_launch(void* const* d_in, const int* in_sizes, int n_in,
                              void* d_out, int out_size, void* d_ws, size_t ws_size,
                              hipStream_t stream) {
  const float* x      = (const float*)d_in[0];
  const float* mask   = (const float*)d_in[1];
  const float* lanm   = (const float*)d_in[2];
  const float* qkv_w  = (const float*)d_in[3];
  const float* qkv_b  = (const float*)d_in[4];
  const float* proj_w = (const float*)d_in[5];
  const float* proj_b = (const float*)d_in[6];
  const int*   lidx   = (const int*)d_in[7];
  float* bm  = (float*)d_ws;           // 64*4*64*64 f32 = 4 MB (L2 resident)
  float* out = (float*)d_out;

  bias_mask_kernel<<<BM_ELEMS / 256, 256, 0, stream>>>(lanm, lidx, mask, bm);
  window_attn_kernel<<<1024, 256, SMEM_BYTES, stream>>>(x, qkv_w, qkv_b,
                                                        proj_w, proj_b, bm, out);
  (void)in_sizes; (void)n_in; (void)out_size; (void)ws_size;
}
